// ImprovedVectorizedCliqueGNN_48790828483124
// MI455X (gfx1250) — matrix operand order repaired and checked by reading, not compiled, not run on hardware
//
#include <hip/hip_runtime.h>

// ---------------------------------------------------------------------------
// CliqueGNN forward for MI455X (gfx1250, wave32, WMMA).
// B=1024, N=32 nodes, H=128, E=496 edges (complete graph K32), L=2 layers.
// All large GEMMs run on v_wmma_f32_16x16x32_f16 with f32 accumulation.
// Weights are pre-swizzled into the WMMA B-fragment lane layout so every
// fragment load is a pair of 16-byte vector loads. Segment-mean over the
// complete graph uses LDS ds_add_f32 atomics (in-degree == 31 everywhere).
// ---------------------------------------------------------------------------

#define BB   1024
#define NN   32
#define HH   128
#define EE   496
#define EE2  992
#define HQQ  32

typedef __attribute__((ext_vector_type(16))) _Float16 v16h;
typedef __attribute__((ext_vector_type(8)))  _Float16 v8h;
typedef __attribute__((ext_vector_type(8)))  float    v8f;

__device__ __forceinline__ v8f wmma_f16(v16h a, v16h b, v8f c) {
  // (neg_a, A, neg_b, B, c_mod, C, reuse_a, reuse_b)
  return __builtin_amdgcn_wmma_f32_16x16x32_f16(false, a, false, b, (short)0, c,
                                                false, false);
}

// 16-bit A-matrix 16x32 lane layout (ISA 7.12.2): lane L holds row M=L&15,
// half=L>>4; elements e<8 -> K = 8*half+e ; e>=8 -> K = 16+8*half+(e-8).
// Both runs are 8 contiguous halfs (16B) -> two vector loads.
__device__ __forceinline__ v16h load_a_row(const _Float16* rowBase, int kofs,
                                           int half) {
  const _Float16* p = rowBase + kofs + 8 * half;
  v8h lo = ((const v8h*)p)[0];        // K = kofs+8h .. +7
  v8h hi = ((const v8h*)(p + 16))[0]; // K = kofs+16+8h .. +7
  v16h a;
#pragma unroll
  for (int e = 0; e < 8; ++e) { a[e] = lo[e]; a[e + 8] = hi[e]; }
  return a;
}

// Swizzled B storage: per 32x16 (KxN) block, lane's 16 halfs contiguous:
//   dst[(((kb*nTiles)+nb)*32 + lane)*16 + e] = W[(kb*32 + (e<8 ? 8*(lane>>4)+e : 16+8*(lane>>4)+e-8))*N + nb*16+(lane&15)]
__device__ __forceinline__ v16h load_b_sw(const _Float16* __restrict__ W,
                                          int nTiles, int kb, int nb, int lane) {
  const _Float16* p = W + ((((size_t)kb * nTiles + nb) * 32 + lane) << 4);
  v8h lo = ((const v8h*)p)[0];
  v8h hi = ((const v8h*)p)[1];
  v16h b;
#pragma unroll
  for (int e = 0; e < 8; ++e) { b[e] = lo[e]; b[e + 8] = hi[e]; }
  return b;
}

// ---------------------------------------------------------------------------
// Weight conversion: f32 [K][N] row-major -> f16 swizzled B-fragment layout.
// ---------------------------------------------------------------------------
__global__ __launch_bounds__(256) void swizzle_w_kernel(const float* __restrict__ src,
                                                        _Float16* __restrict__ dst,
                                                        int K, int Ncols) {
  int i = blockIdx.x * 256 + threadIdx.x;
  if (i >= K * Ncols) return;
  int e = i & 15;
  int lane = (i >> 4) & 31;
  int blk = i >> 9;                 // kb*(N/16) + nb
  int nT = Ncols >> 4;
  int kb = blk / nT, nb = blk - kb * nT;
  int n = nb * 16 + (lane & 15);
  int k = kb * 32 + ((e < 8) ? 8 * (lane >> 4) + e : 16 + 8 * (lane >> 4) + (e - 8));
  dst[i] = (_Float16)src[k * Ncols + n];
}

// nodes = zeros @ Wn + bn  ==  broadcast(bn)
__global__ __launch_bounds__(256) void init_nodes_kernel(float* __restrict__ nodes,
                                                         const float* __restrict__ bn) {
  size_t i = (size_t)blockIdx.x * 256 + threadIdx.x;
  if (i < (size_t)BB * NN * HH) nodes[i] = bn[i & (HH - 1)];
}

// ef = edge_features(B,E,3) @ We(3,H) + be
__global__ __launch_bounds__(256) void init_ef_kernel(const float* __restrict__ efeat,
                                                      const float* __restrict__ We,
                                                      const float* __restrict__ be,
                                                      float* __restrict__ ef) {
  size_t i = (size_t)blockIdx.x * 256 + threadIdx.x;
  if (i >= (size_t)BB * EE * HH) return;
  int h = (int)(i & (HH - 1));
  size_t row = i >> 7;
  const float* fr = efeat + row * 3;
  ef[i] = be[h] + fr[0] * We[0 * HH + h] + fr[1] * We[1 * HH + h] + fr[2] * We[2 * HH + h];
}

// ---------------------------------------------------------------------------
// GNN block: msg = relu([nodes[src], ef] @ msg_W + b) over 2E directed edges,
// agg = segment_mean(msg, dst) (cnt==31 on K32), nodes = LN(agg + nodes).
// One workgroup (8 waves) per batch element; msg_W swizzled in LDS.
// ---------------------------------------------------------------------------
__global__ __launch_bounds__(256) void gnn_block_kernel(
    const int* __restrict__ ei,           // batch0 (2,E): ei[e]=i, ei[E+e]=j
    float* __restrict__ nodes,            // (B,N,H) in/out
    const float* __restrict__ ef,         // (B,E,H)
    const _Float16* __restrict__ msgW16,  // (2H,H) f16, swizzled
    const float* __restrict__ msg_b,
    const float* __restrict__ lnn_s, const float* __restrict__ lnn_b) {
  __shared__ _Float16 sNodesH[NN * HH];     // 8 KB
  __shared__ float    sNodesF[NN * HH];     // 16 KB (residual)
  __shared__ float    sAgg[NN * HH];        // 16 KB (atomic accumulator)
  __shared__ _Float16 sW[256 * HH];         // 64 KB msg_W (swizzled)
  __shared__ float    sBias[HH];
  __shared__ _Float16 sEft[8][16 * HH];     // per-wave ef tile staging, 32 KB

  int b = blockIdx.x;
  int tid = threadIdx.x, lane = tid & 31, wave = tid >> 5;

  float* nb = nodes + (size_t)b * NN * HH;
  for (int i = tid; i < NN * HH; i += 256) {
    float v = nb[i];
    sNodesF[i] = v;
    sNodesH[i] = (_Float16)v;
    sAgg[i] = 0.0f;
  }
  for (int i = tid; i < 256 * HH / 8; i += 256)       // verbatim copy, 16B chunks
    ((v8h*)sW)[i] = ((const v8h*)msgW16)[i];
  if (tid < HH) sBias[tid] = msg_b[tid];
  __syncthreads();

  const float* efb = ef + (size_t)b * EE * HH;
  int mA = lane & 15, half = lane >> 4;

  for (int t = wave; t < EE2 / 16; t += 8) {          // 62 row tiles of 16
    int r0 = t * 16;
    _Float16* eft = sEft[wave];
    // stage ef tile (f32 -> f16), float4 vectorized
    for (int i = lane; i < 16 * (HH / 4); i += 32) {
      int rr = i >> 5, c4 = i & 31;
      int d = r0 + rr;
      int eidx = (d < EE) ? d : d - EE;
      const float4 v = ((const float4*)(efb + (size_t)eidx * HH))[c4];
      _Float16* dh = eft + rr * HH + c4 * 4;
      dh[0] = (_Float16)v.x; dh[1] = (_Float16)v.y;
      dh[2] = (_Float16)v.z; dh[3] = (_Float16)v.w;
    }
    // src = concat(ei0,ei1)[d] == ei[d];  dst = concat(ei1,ei0)[d]
    int srcn = ei[r0 + mA];
    const _Float16* aSrcRow = sNodesH + srcn * HH;
    const _Float16* aEfRow = eft + mA * HH;
    int dst8[8];
#pragma unroll
    for (int i = 0; i < 8; ++i) {
      int d = r0 + i + 8 * half;
      dst8[i] = (d < EE) ? ei[EE + d] : ei[d - EE];
    }
    for (int ct = 0; ct < 8; ++ct) {                   // 128 out cols
      v8f c = {};
#pragma unroll
      for (int ks = 0; ks < 8; ++ks) {                 // K = 256
        v16h a = (ks < 4) ? load_a_row(aSrcRow, ks * 32, half)
                          : load_a_row(aEfRow, ks * 32 - HH, half);
        v16h bf = load_b_sw(sW, 8, ks, ct, lane);
        c = wmma_f16(a, bf, c);
      }
      int n = ct * 16 + (lane & 15);
      float bias = sBias[n];
#pragma unroll
      for (int i = 0; i < 8; ++i) {
        float v = c[i] + bias;
        v = v > 0.0f ? v : 0.0f;
        atomicAdd(&sAgg[dst8[i] * HH + n], v);         // ds_add_f32
      }
    }
  }
  __syncthreads();

  // LayerNorm(agg/31 + nodes); one thread per node row (trivial vs GEMM cost)
  if (tid < NN) {
    int r = tid;
    const float inv31 = 1.0f / 31.0f;
    float mean = 0.0f;
    for (int h = 0; h < HH; ++h) mean += sAgg[r * HH + h] * inv31 + sNodesF[r * HH + h];
    mean *= (1.0f / HH);
    float var = 0.0f;
    for (int h = 0; h < HH; ++h) {
      float v = sAgg[r * HH + h] * inv31 + sNodesF[r * HH + h] - mean;
      var += v * v;
    }
    var *= (1.0f / HH);
    float inv = rsqrtf(var + 1e-6f);
    for (int h = 0; h < HH; ++h) {
      float v = sAgg[r * HH + h] * inv31 + sNodesF[r * HH + h] - mean;
      nb[r * HH + h] = lnn_s[h] * v * inv + lnn_b[h];
    }
  }
}

// ---------------------------------------------------------------------------
// Edge block: fused ep / np1 / np2 / o1 / o2 / LayerNorm per 16-edge tile.
// One workgroup (4 waves) per batch; ef read once, written once.
// Weights read from L2-resident swizzled f16 buffers (2x b128 per fragment).
// ---------------------------------------------------------------------------
__global__ __launch_bounds__(128) void edge_block_kernel(
    const int* __restrict__ ei, const float* __restrict__ nodes,
    float* __restrict__ ef,
    const _Float16* __restrict__ npW16, const _Float16* __restrict__ epW16,
    const _Float16* __restrict__ cbW16,
    const float* __restrict__ np_b, const float* __restrict__ ep_b,
    const float* __restrict__ cb_b,
    const float* __restrict__ lne_s, const float* __restrict__ lne_b) {
  __shared__ _Float16 sNodesH[NN * HH];     // 8 KB
  __shared__ float sNpB[HH], sEpB[HH], sCbB[HH];
  __shared__ _Float16 sNp[4][16 * HH];      // 16 KB (np1/np2, reused per pass)
  __shared__ _Float16 sEp[4][16 * HH];      // 16 KB
  __shared__ _Float16 sEfh[4][16 * HH];     // 16 KB (ef tile f16 for WMMA A)
  __shared__ float    sEft[4][16 * HH];     // 32 KB (residual f32)
  __shared__ float    sO[4][16 * HH];       // 32 KB (0.5*(o1+o2))

  int b = blockIdx.x, tid = threadIdx.x, lane = tid & 31, wave = tid >> 5;
  const float* nbase = nodes + (size_t)b * NN * HH;
  float* efb = ef + (size_t)b * EE * HH;

  for (int i = tid; i < NN * HH; i += 128) sNodesH[i] = (_Float16)nbase[i];
  if (tid < HH) { sNpB[tid] = np_b[tid]; sEpB[tid] = ep_b[tid]; sCbB[tid] = cb_b[tid]; }
  __syncthreads();

  int mA = lane & 15, half = lane >> 4, nLo = lane & 15;

  for (int t = wave; t < EE / 16; t += 4) {            // 31 edge tiles
    int r0 = t * 16;
    float*    eft = sEft[wave];
    _Float16* efh = sEfh[wave];
    _Float16* npS = sNp[wave];
    _Float16* epS = sEp[wave];
    float*    oS  = sO[wave];

    for (int i = lane; i < 16 * (HH / 4); i += 32) {   // stage ef tile f32+f16
      int rr = i >> 5, c4 = i & 31;
      const float4 v = ((const float4*)(efb + (size_t)(r0 + rr) * HH))[c4];
      float* d = eft + rr * HH + c4 * 4;
      d[0] = v.x; d[1] = v.y; d[2] = v.z; d[3] = v.w;
      _Float16* dh = efh + rr * HH + c4 * 4;
      dh[0] = (_Float16)v.x; dh[1] = (_Float16)v.y;
      dh[2] = (_Float16)v.z; dh[3] = (_Float16)v.w;
    }
    int sNode = ei[r0 + mA];
    int dNode = ei[EE + r0 + mA];
    const _Float16* aS = sNodesH + sNode * HH;
    const _Float16* aD = sNodesH + dNode * HH;
    const _Float16* aEf = efh + mA * HH;
    const _Float16* aNp = npS + mA * HH;
    const _Float16* aEp = epS + mA * HH;

    // ep = relu(ef @ ep_W + ep_b), K=128
    for (int ct = 0; ct < 8; ++ct) {
      v8f c = {};
#pragma unroll
      for (int ks = 0; ks < 4; ++ks) {
        v16h a = load_a_row(aEf, ks * 32, half);
        v16h bf = load_b_sw(epW16, 8, ks, ct, lane);
        c = wmma_f16(a, bf, c);
      }
      int n = ct * 16 + nLo;
      float bias = sEpB[n];
#pragma unroll
      for (int i = 0; i < 8; ++i) {
        float v = c[i] + bias; v = v > 0.0f ? v : 0.0f;
        epS[(i + 8 * half) * HH + n] = (_Float16)v;
      }
    }

    // pass 0: np1 = relu([sf,df]@np_W), o1;  pass 1: np2 = relu([df,sf]@np_W), o2
    for (int pass = 0; pass < 2; ++pass) {
      const _Float16* aLo = (pass == 0) ? aS : aD;
      const _Float16* aHi = (pass == 0) ? aD : aS;
      for (int ct = 0; ct < 8; ++ct) {                 // np GEMM, K=256
        v8f c = {};
#pragma unroll
        for (int ks = 0; ks < 8; ++ks) {
          v16h a = (ks < 4) ? load_a_row(aLo, ks * 32, half)
                            : load_a_row(aHi, ks * 32 - HH, half);
          v16h bf = load_b_sw(npW16, 8, ks, ct, lane);
          c = wmma_f16(a, bf, c);
        }
        int n = ct * 16 + nLo;
        float bias = sNpB[n];
#pragma unroll
        for (int i = 0; i < 8; ++i) {
          float v = c[i] + bias; v = v > 0.0f ? v : 0.0f;
          npS[(i + 8 * half) * HH + n] = (_Float16)v;
        }
      }
      for (int ct = 0; ct < 8; ++ct) {                 // o GEMM, K=256
        v8f c = {};
#pragma unroll
        for (int ks = 0; ks < 8; ++ks) {
          v16h a = (ks < 4) ? load_a_row(aNp, ks * 32, half)
                            : load_a_row(aEp, ks * 32 - HH, half);
          v16h bf = load_b_sw(cbW16, 8, ks, ct, lane);
          c = wmma_f16(a, bf, c);
        }
        int n = ct * 16 + nLo;
        float bias = sCbB[n];
#pragma unroll
        for (int i = 0; i < 8; ++i) {
          float v = c[i] + bias; v = v > 0.0f ? v : 0.0f;
          int idx = (i + 8 * half) * HH + n;
          if (pass == 0) oS[idx] = 0.5f * v; else oS[idx] += 0.5f * v;
        }
      }
    }

    // LayerNorm(0.5*(o1+o2) + ef) per row
    if (lane < 16) {
      int m = lane;
      float mean = 0.0f;
      for (int h = 0; h < HH; ++h) mean += oS[m * HH + h] + eft[m * HH + h];
      mean *= (1.0f / HH);
      float var = 0.0f;
      for (int h = 0; h < HH; ++h) {
        float v = oS[m * HH + h] + eft[m * HH + h] - mean;
        var += v * v;
      }
      var *= (1.0f / HH);
      float inv = rsqrtf(var + 1e-6f);
      for (int h = 0; h < HH; ++h) {
        float v = oS[m * HH + h] + eft[m * HH + h] - mean;
        efb[(size_t)(r0 + m) * HH + h] = lne_s[h] * v * inv + lne_b[h];
      }
    }
  }
}

// ---------------------------------------------------------------------------
// Policy head: x1 = relu(ef@pW1), x2 = relu(x1@pW2), logits = x2@pW3 + pb3,
// softmax over E. One workgroup (4 waves) per batch element.
// ---------------------------------------------------------------------------
__global__ __launch_bounds__(128) void policy_kernel(
    const float* __restrict__ ef,
    const _Float16* __restrict__ pW1h, const _Float16* __restrict__ pW2h,
    const float* __restrict__ pW3, const float* __restrict__ pb1,
    const float* __restrict__ pb2, const float* __restrict__ pb3,
    float* __restrict__ out_pol) {
  __shared__ _Float16 sEfh[4][16 * HH];     // 16 KB
  __shared__ _Float16 sX1[4][16 * 256];     // 32 KB
  __shared__ float    sX2[4][16 * HH];      // 32 KB
  __shared__ float sB1[256], sB2[HH], sW3[HH];
  __shared__ float sLogits[EE];
  __shared__ float sMax, sSum;

  int b = blockIdx.x, tid = threadIdx.x, lane = tid & 31, wave = tid >> 5;
  const float* efb = ef + (size_t)b * EE * HH;
  for (int i = tid; i < 256; i += 128) sB1[i] = pb1[i];
  if (tid < HH) { sB2[tid] = pb2[tid]; sW3[tid] = pW3[tid]; }
  __syncthreads();

  int mA = lane & 15, half = lane >> 4, nLo = lane & 15;

  for (int t = wave; t < EE / 16; t += 4) {
    int r0 = t * 16;
    _Float16* efh = sEfh[wave];
    _Float16* x1  = sX1[wave];
    float*    x2  = sX2[wave];
    for (int i = lane; i < 16 * (HH / 4); i += 32) {
      int rr = i >> 5, c4 = i & 31;
      const float4 v = ((const float4*)(efb + (size_t)(r0 + rr) * HH))[c4];
      _Float16* d = efh + rr * HH + c4 * 4;
      d[0] = (_Float16)v.x; d[1] = (_Float16)v.y;
      d[2] = (_Float16)v.z; d[3] = (_Float16)v.w;
    }
    const _Float16* aEf = efh + mA * HH;
    const _Float16* aX1 = x1 + mA * 256;
    for (int ct = 0; ct < 16; ++ct) {                  // x1: 256 cols, K=128
      v8f c = {};
#pragma unroll
      for (int ks = 0; ks < 4; ++ks) {
        v16h a = load_a_row(aEf, ks * 32, half);
        v16h bf = load_b_sw(pW1h, 16, ks, ct, lane);
        c = wmma_f16(a, bf, c);
      }
      int n = ct * 16 + nLo;
      float bias = sB1[n];
#pragma unroll
      for (int i = 0; i < 8; ++i) {
        float v = c[i] + bias; v = v > 0.0f ? v : 0.0f;
        x1[(i + 8 * half) * 256 + n] = (_Float16)v;
      }
    }
    for (int ct = 0; ct < 8; ++ct) {                   // x2: 128 cols, K=256
      v8f c = {};
#pragma unroll
      for (int ks = 0; ks < 8; ++ks) {
        v16h a = load_a_row(aX1, ks * 32, half);
        v16h bf = load_b_sw(pW2h, 8, ks, ct, lane);
        c = wmma_f16(a, bf, c);
      }
      int n = ct * 16 + nLo;
      float bias = sB2[n];
#pragma unroll
      for (int i = 0; i < 8; ++i) {
        float v = c[i] + bias;
        x2[(i + 8 * half) * HH + n] = v > 0.0f ? v : 0.0f;
      }
    }
    if (lane < 16) {
      float acc = pb3[0];
      for (int h = 0; h < HH; ++h) acc += x2[lane * HH + h] * sW3[h];
      sLogits[r0 + lane] = acc;
    }
  }
  __syncthreads();
  if (tid == 0) {
    float mx = -1e30f;
    for (int i = 0; i < EE; ++i) mx = fmaxf(mx, sLogits[i]);
    float sm = 0.0f;
    for (int i = 0; i < EE; ++i) sm += expf(sLogits[i] - mx);
    sMax = mx; sSum = sm;
  }
  __syncthreads();
  float inv = 1.0f / sSum;
  for (int i = tid; i < EE; i += 128)
    out_pol[(size_t)b * EE + i] = expf(sLogits[i] - sMax) * inv;
}

// ---------------------------------------------------------------------------
// Attention pooling + batchnorm + value MLP (≈1% of FLOPs, plain VALU).
// ---------------------------------------------------------------------------
__global__ __launch_bounds__(256) void value_kernel(
    const float* __restrict__ nodes, const float* __restrict__ ef,
    const float* __restrict__ naW1, const float* __restrict__ nab1,
    const float* __restrict__ naW2, const float* __restrict__ nab2,
    const float* __restrict__ eaW1, const float* __restrict__ eab1,
    const float* __restrict__ eaW2, const float* __restrict__ eab2,
    const float* __restrict__ bn_s, const float* __restrict__ bn_b,
    const float* __restrict__ bn_m, const float* __restrict__ bn_v,
    const float* __restrict__ vW1, const float* __restrict__ vb1,
    const float* __restrict__ vW2, const float* __restrict__ vb2,
    const float* __restrict__ vW3, const float* __restrict__ vb3,
    float* __restrict__ out_val) {
  __shared__ float sNodes[NN * HH];         // 16 KB
  __shared__ float sAW1[HH * HQQ];          // 16 KB (naW1 then eaW1)
  __shared__ float sAW2[HQQ];
  __shared__ float sScores[EE];
  __shared__ float sPart[NN][8];
  __shared__ float sPool[2 * HH];
  __shared__ float sG[2 * HH];
  __shared__ float sV1[HH], sV2[HH];

  int b = blockIdx.x, tid = threadIdx.x;
  const float* nbase = nodes + (size_t)b * NN * HH;
  const float* efb = ef + (size_t)b * EE * HH;

  for (int i = tid; i < NN * HH; i += 256) sNodes[i] = nbase[i];
  for (int i = tid; i < HH * HQQ; i += 256) sAW1[i] = naW1[i];
  if (tid < HQQ) sAW2[tid] = naW2[tid];
  __syncthreads();

  {  // node attention scores: 32 rows x 8 threads, 4 j-cols per thread
    int r = tid >> 3, g = tid & 7;
    float part = 0.0f;
    for (int jj = 0; jj < 4; ++jj) {
      int j = g * 4 + jj;
      float tacc = nab1[j];
      for (int k = 0; k < HH; ++k) tacc += sNodes[r * HH + k] * sAW1[k * HQQ + j];
      part += tanhf(tacc) * sAW2[j];
    }
    sPart[r][g] = part;
  }
  __syncthreads();
  if (tid < NN) {
    float s = nab2[0];
    for (int g = 0; g < 8; ++g) s += sPart[tid][g];
    sScores[tid] = s;
  }
  __syncthreads();
  if (tid == 0) {
    float mx = -1e30f;
    for (int i = 0; i < NN; ++i) mx = fmaxf(mx, sScores[i]);
    float sm = 0.0f;
    for (int i = 0; i < NN; ++i) sm += expf(sScores[i] - mx);
    for (int i = 0; i < NN; ++i) sScores[i] = expf(sScores[i] - mx) / sm;
  }
  __syncthreads();
  if (tid < HH) {
    float acc = 0.0f;
    for (int r = 0; r < NN; ++r) acc += sNodes[r * HH + tid] * sScores[r];
    sPool[tid] = acc;
  }
  __syncthreads();

  for (int i = tid; i < HH * HQQ; i += 256) sAW1[i] = eaW1[i];   // restage eaW1
  if (tid < HQQ) sAW2[tid] = eaW2[tid];
  __syncthreads();

  for (int r = tid; r < EE; r += 256) {     // edge attention scores
    float acc32[HQQ];
#pragma unroll
    for (int j = 0; j < HQQ; ++j) acc32[j] = eab1[j];
    const float* fr = efb + (size_t)r * HH;
    for (int k = 0; k < HH; ++k) {
      float f = fr[k];
#pragma unroll
      for (int j = 0; j < HQQ; ++j) acc32[j] += f * sAW1[k * HQQ + j];
    }
    float s = eab2[0];
#pragma unroll
    for (int j = 0; j < HQQ; ++j) s += tanhf(acc32[j]) * sAW2[j];
    sScores[r] = s;
  }
  __syncthreads();
  if (tid == 0) {
    float mx = -1e30f;
    for (int i = 0; i < EE; ++i) mx = fmaxf(mx, sScores[i]);
    float sm = 0.0f;
    for (int i = 0; i < EE; ++i) sm += expf(sScores[i] - mx);
    for (int i = 0; i < EE; ++i) sScores[i] = expf(sScores[i] - mx) / sm;
  }
  __syncthreads();
  if (tid < HH) {
    float acc = 0.0f;
    for (int r = 0; r < EE; ++r) acc += efb[(size_t)r * HH + tid] * sScores[r];
    sPool[HH + tid] = acc;
  }
  __syncthreads();
  if (tid < 2 * HH)
    sG[tid] = (sPool[tid] - bn_m[tid]) * rsqrtf(bn_v[tid] + 1e-5f) * bn_s[tid] + bn_b[tid];
  __syncthreads();
  if (tid < HH) {
    float acc = vb1[tid];
    for (int i = 0; i < 2 * HH; ++i) acc += sG[i] * vW1[i * HH + tid];
    sV1[tid] = fmaxf(acc, 0.0f);
  }
  __syncthreads();
  if (tid < HH) {
    float acc = vb2[tid];
    for (int k = 0; k < HH; ++k) acc += sV1[k] * vW2[k * HH + tid];
    sV2[tid] = fmaxf(acc, 0.0f);
  }
  __syncthreads();
  if (tid == 0) {
    float acc = vb3[0];
    for (int h = 0; h < HH; ++h) acc += sV2[h] * vW3[h];
    out_val[b] = tanhf(acc);
  }
}

// ---------------------------------------------------------------------------
// Host launcher. Input order: edge_index, edge_features, then params flattened
// pytree-style (dict keys sorted ASCII; 'layers' list in order, each layer
// dict keys sorted: cb_W, cb_b, ep_W, ep_b, lne_b, lne_s, lnn_b, lnn_s,
// msg_W, msg_b, np_W, np_b).
// ---------------------------------------------------------------------------
extern "C" void kernel_launch(void* const* d_in, const int* in_sizes, int n_in,
                              void* d_out, int out_size, void* d_ws, size_t ws_size,
                              hipStream_t stream) {
  (void)in_sizes; (void)n_in; (void)out_size; (void)ws_size;
  const int* edge_index = (const int*)d_in[0];        // (B,2,E); batch0 used
  const float* edge_features = (const float*)d_in[1]; // (B,E,3)
  int p = 2;
  auto F = [&]() { return (const float*)d_in[p++]; };
  // sorted: We, Wn, be, bn, bn_b, bn_m, bn_s, bn_v, eaW1, eaW2, eab1, eab2
  const float* We = F();  const float* Wn = F(); (void)Wn;
  const float* be = F();  const float* bn = F();
  const float* bn_b = F(); const float* bn_m = F();
  const float* bn_s = F(); const float* bn_v = F();
  const float* eaW1 = F(); const float* eaW2 = F();
  const float* eab1 = F(); const float* eab2 = F();
  struct LayerP {
    const float *cb_W, *cb_b, *ep_W, *ep_b, *lne_b, *lne_s, *lnn_b, *lnn_s,
                *msg_W, *msg_b, *np_W, *np_b;
  } L[2];
  for (int l = 0; l < 2; ++l) {
    L[l].cb_W = F();  L[l].cb_b = F();  L[l].ep_W = F();  L[l].ep_b = F();
    L[l].lne_b = F(); L[l].lne_s = F(); L[l].lnn_b = F(); L[l].lnn_s = F();
    L[l].msg_W = F(); L[l].msg_b = F(); L[l].np_W = F();  L[l].np_b = F();
  }
  const float* naW1 = F(); const float* naW2 = F();
  const float* nab1 = F(); const float* nab2 = F();
  const float* pW1 = F(); const float* pW2 = F(); const float* pW3 = F();
  const float* pb1 = F(); const float* pb2 = F(); const float* pb3 = F();
  const float* vW1 = F(); const float* vW2 = F(); const float* vW3 = F();
  const float* vb1 = F(); const float* vb2 = F(); const float* vb3 = F();

  // workspace carve-up
  char* ws = (char*)d_ws;
  float* efBuf = (float*)ws;    ws += (size_t)BB * EE * HH * sizeof(float);
  float* nodesBuf = (float*)ws; ws += (size_t)BB * NN * HH * sizeof(float);
  _Float16* w16 = (_Float16*)ws;
  size_t off = 0;
  _Float16 *msgW16[2], *npW16[2], *cbW16[2], *epW16[2];
  for (int l = 0; l < 2; ++l) {
    msgW16[l] = w16 + off; off += 256 * HH;
    npW16[l]  = w16 + off; off += 256 * HH;
    cbW16[l]  = w16 + off; off += 256 * HH;
    epW16[l]  = w16 + off; off += HH * HH;
  }
  _Float16* pW1h = w16 + off; off += HH * 256;
  _Float16* pW2h = w16 + off; off += 256 * HH;

  auto swz = [&](const float* src, _Float16* dst, int K, int Ncols) {
    int n = K * Ncols;
    swizzle_w_kernel<<<(n + 255) / 256, 256, 0, stream>>>(src, dst, K, Ncols);
  };
  for (int l = 0; l < 2; ++l) {
    swz(L[l].msg_W, msgW16[l], 256, HH);
    swz(L[l].np_W, npW16[l], 256, HH);
    swz(L[l].cb_W, cbW16[l], 256, HH);
    swz(L[l].ep_W, epW16[l], HH, HH);
  }
  swz(pW1, pW1h, HH, 256);
  swz(pW2, pW2h, 256, HH);

  {
    size_t n = (size_t)BB * NN * HH;
    init_nodes_kernel<<<(unsigned)((n + 255) / 256), 256, 0, stream>>>(nodesBuf, bn);
  }
  {
    size_t n = (size_t)BB * EE * HH;
    init_ef_kernel<<<(unsigned)((n + 255) / 256), 256, 0, stream>>>(edge_features, We,
                                                                    be, efBuf);
  }
  for (int l = 0; l < 2; ++l) {
    gnn_block_kernel<<<BB, 256, 0, stream>>>(edge_index, nodesBuf, efBuf, msgW16[l],
                                             L[l].msg_b, L[l].lnn_s, L[l].lnn_b);
    edge_block_kernel<<<BB, 128, 0, stream>>>(edge_index, nodesBuf, efBuf, npW16[l],
                                              epW16[l], cbW16[l], L[l].np_b, L[l].ep_b,
                                              L[l].cb_b, L[l].lne_s, L[l].lne_b);
  }
  policy_kernel<<<BB, 128, 0, stream>>>(efBuf, pW1h, pW2h, pW3, pb1, pb2, pb3,
                                        (float*)d_out);
  value_kernel<<<BB, 256, 0, stream>>>(nodesBuf, efBuf, naW1, nab1, naW2, nab2, eaW1,
                                       eab1, eaW2, eab2, bn_s, bn_b, bn_m, bn_v, vW1,
                                       vb1, vW2, vb2, vW3, vb3,
                                       (float*)d_out + (size_t)BB * EE);
}
